// DurationConditioningProjector_52321291600409
// MI455X (gfx1250) — matrix operand order repaired and checked by reading, not compiled, not run on hardware
//
#include <hip/hip_runtime.h>
#include <hip/hip_bf16.h>
#include <math.h>

// Problem constants (match reference)
#define B_   8
#define N_   1024
#define DIN_ 512
#define C_   256
#define T_   8192
#define K_   31
#define L_   3
#define EPS_ 1e-5f
#define TM   16      // time-rows per stage1 GEMM block
#define TMC  32      // time-rows per conv GEMM block (2 M-tiles per wave)

typedef __attribute__((ext_vector_type(16))) __bf16 v16bf;
typedef __attribute__((ext_vector_type(8)))  float  v8f;

// ---------- helpers ----------
__device__ __forceinline__ __bf16 f2bf(float f) {
  union { float f; unsigned u; } v; v.f = f;
  unsigned r = v.u + 0x7FFFu + ((v.u >> 16) & 1u);   // round-nearest-even
  unsigned short h = (unsigned short)(r >> 16);
  __bf16 o; __builtin_memcpy(&o, &h, 2);
  return o;
}

__device__ __forceinline__ v8f wmma_bf16(v16bf a, v16bf b, v8f c) {
  return __builtin_amdgcn_wmma_f32_16x16x32_bf16(
      /*neg_a=*/false, a, /*neg_b=*/false, b,
      /*c_mod=*/(short)0, c, /*reuse_a=*/false, /*reuse_b=*/false);
}

// A fragment (16x32 bf16, MxK) from LDS, rows strided by ldm elements.
__device__ __forceinline__ v16bf load_a_frag(const __bf16* p, int ldm) {
  int lane = threadIdx.x & 31;
  int half = lane >> 4;
  int m    = lane & 15;
  const __bf16* row = p + m * ldm;
  v16bf a;
#pragma unroll
  for (int i = 0; i < 4; ++i) {
    int k0 = half * 16 + 2 * i;
    a[2 * i]     = row[k0];
    a[2 * i + 1] = row[k0 + 1];
    int k1 = half * 16 + 8 + 2 * i;
    a[2 * (i + 4)]     = row[k1];
    a[2 * (i + 4) + 1] = row[k1 + 1];
  }
  return a;
}

// B fragment (32x16 bf16, KxN) pre-packed fragment-linear: lane's 16 elems contiguous (32B)
__device__ __forceinline__ v16bf load_b_frag(const __bf16* tile_base) {
  int lane = threadIdx.x & 31;
  return *((const v16bf*)(tile_base) + lane);   // 2x global_load_b128
}

// ---------- kernel 0: per-batch cumsum of durations ----------
__global__ void k_cumsum(const int* __restrict__ dur, int* __restrict__ cum) {
  int b = blockIdx.x;
  if (threadIdx.x == 0) {
    int s = 0;
    for (int n = 0; n < N_; ++n) { s += dur[b * N_ + n]; cum[b * N_ + n] = s; }
  }
}

// ---------- kernel 1: frame -> phoneme index (lower_bound cum[n] > t) ----------
__global__ void k_index(const int* __restrict__ cum, int* __restrict__ idx) {
  int i = blockIdx.x * blockDim.x + threadIdx.x;
  if (i >= B_ * T_) return;
  int b = i / T_, t = i % T_;
  const int* c = cum + b * N_;
  int lo = 0, hi = N_ - 1;
  while (lo < hi) { int mid = (lo + hi) >> 1; if (c[mid] > t) hi = mid; else lo = mid + 1; }
  idx[i] = lo;
}

// ---------- kernel 2: pack plain [Kdim x Ndim] fp32 matrix -> bf16 B fragments ----------
__global__ void k_pack(const float* __restrict__ W, __bf16* __restrict__ out,
                       int Kdim, int Ndim) {
  int total = Kdim * Ndim;
  int NT = Ndim >> 4;
  for (int i = blockIdx.x * blockDim.x + threadIdx.x; i < total;
       i += gridDim.x * blockDim.x) {
    int e    = i & 15;
    int lane = (i >> 4) & 31;
    int tile = i >> 9;
    int kt = tile / NT, nt = tile % NT;
    int k = kt * 32 + (lane >> 4) * 16 + e;
    int n = nt * 16 + (lane & 15);
    out[i] = f2bf(W[k * Ndim + n]);
  }
}

// ---------- kernel 3: pack conv weights [L][Co][Ci][K] -> per-(l,tap) bf16 B fragments ----------
__global__ void k_pack_conv(const float* __restrict__ cw, __bf16* __restrict__ out) {
  const int perLK = C_ * C_;               // 65536 frag elems per (l,tap)
  int total = L_ * K_ * perLK;
  for (int i = blockIdx.x * blockDim.x + threadIdx.x; i < total;
       i += gridDim.x * blockDim.x) {
    int lk    = i / perLK;
    int local = i % perLK;
    int l = lk / K_, k = lk % K_;
    int e    = local & 15;
    int lane = (local >> 4) & 31;
    int tile = local >> 9;                  // 128 tiles: kt*16 + nt
    int kt = tile >> 4, nt = tile & 15;
    int ci = kt * 32 + (lane >> 4) * 16 + e;
    int co = nt * 16 + (lane & 15);
    out[i] = f2bf(cw[(((size_t)l * C_ + co) * C_ + ci) * K_ + k]);
  }
}

// ---------- kernel 4: upsample-gather + in_proj + pos_emb@W_pos  (WMMA) ----------
__global__ __launch_bounds__(256) void k_stage1(
    const float* __restrict__ pooled, const int* __restrict__ idx,
    const float* __restrict__ rel_pos,
    const __bf16* __restrict__ Win_f, const float* __restrict__ b_in,
    const __bf16* __restrict__ Wpos_f, const float* __restrict__ b_pos,
    float* __restrict__ X) {
  const int b  = blockIdx.y;
  const int t0 = blockIdx.x * TM;
  __shared__ __bf16 As[TM][DIN_ + 8];     // gathered pooled rows, bf16
  __shared__ __bf16 Ps[TM][C_ + 8];       // sinusoidal pos-emb tile, bf16
  __shared__ int   sidx[TM];
  __shared__ float srel[TM];

  int tid = threadIdx.x;
  if (tid < TM)                sidx[tid]      = idx[b * T_ + t0 + tid];
  else if (tid < 2 * TM)       srel[tid - TM] = rel_pos[b * T_ + t0 + (tid - TM)];
  __syncthreads();

  // gather + convert pooled rows (float4 loads)
  const float4* p4 = (const float4*)pooled;
  for (int i = tid; i < TM * (DIN_ / 4); i += 256) {
    int m = i / (DIN_ / 4), d4 = i % (DIN_ / 4);
    float4 v = p4[((size_t)b * N_ + sidx[m]) * (DIN_ / 4) + d4];
    __bf16* dst = &As[m][d4 * 4];
    dst[0] = f2bf(v.x); dst[1] = f2bf(v.y); dst[2] = f2bf(v.z); dst[3] = f2bf(v.w);
  }
  // sinusoidal positional embedding tile
  for (int i = tid; i < TM * (C_ / 4); i += 256) {
    int m = i / (C_ / 4), c4 = (i % (C_ / 4)) * 4;
    float x = srel[m];
#pragma unroll
    for (int j = 0; j < 4; ++j) {
      int c = c4 + j;
      int f = (c < 128) ? c : (c - 128);
      float fr  = __expf(-9.210340371976184f * (float)f / 127.0f);
      float arg = x * fr;
      Ps[m][c] = f2bf((c < 128) ? sinf(arg) : cosf(arg));
    }
  }
  __syncthreads();

  int wave = tid >> 5;
  int lane = tid & 31;
  int half = lane >> 4;
  int nt0  = wave * 2;
  v8f acc0 = {}, acc1 = {};

  // in_proj: K = 512
#pragma unroll 4
  for (int kt = 0; kt < DIN_ / 32; ++kt) {
    v16bf a  = load_a_frag(&As[0][kt * 32], DIN_ + 8);
    v16bf b0 = load_b_frag(Win_f + ((size_t)(kt * 16 + nt0)) * 512);
    v16bf b1 = load_b_frag(Win_f + ((size_t)(kt * 16 + nt0 + 1)) * 512);
    acc0 = wmma_bf16(a, b0, acc0);
    acc1 = wmma_bf16(a, b1, acc1);
  }
  // pos_proj: K = 256
#pragma unroll 4
  for (int kt = 0; kt < C_ / 32; ++kt) {
    v16bf a  = load_a_frag(&Ps[0][kt * 32], C_ + 8);
    v16bf b0 = load_b_frag(Wpos_f + ((size_t)(kt * 16 + nt0)) * 512);
    v16bf b1 = load_b_frag(Wpos_f + ((size_t)(kt * 16 + nt0 + 1)) * 512);
    acc0 = wmma_bf16(a, b0, acc0);
    acc1 = wmma_bf16(a, b1, acc1);
  }

  int n0 = nt0 * 16 + (lane & 15);
  int n1 = n0 + 16;
  float bias0 = b_in[n0] + b_pos[n0];
  float bias1 = b_in[n1] + b_pos[n1];
#pragma unroll
  for (int r = 0; r < 8; ++r) {
    int m = half * 8 + r;
    size_t o = ((size_t)b * T_ + t0 + m) * C_;
    X[o + n0] = acc0[r] + bias0;
    X[o + n1] = acc1[r] + bias1;
  }
}

// ---------- reductions ----------
__device__ __forceinline__ float wave_sum(float v) {
#pragma unroll
  for (int off = 16; off > 0; off >>= 1) v += __shfl_down(v, off, 32);
  return v;
}

// ---------- kernel 5: LayerNorm + GELU -> bf16 h ----------
__global__ __launch_bounds__(256) void k_ln_gelu(
    const float* __restrict__ X, const float* __restrict__ g,
    const float* __restrict__ bt, __bf16* __restrict__ H) {
  int row = blockIdx.x;                       // b*T + t
  int c = threadIdx.x;
  float x = X[(size_t)row * C_ + c];
  __shared__ float red[8];

  float s = wave_sum(x);
  if ((c & 31) == 0) red[c >> 5] = s;
  __syncthreads();
  float mu = (red[0] + red[1] + red[2] + red[3] + red[4] + red[5] + red[6] + red[7]) * (1.0f / C_);
  __syncthreads();
  float d = x - mu;
  float v = wave_sum(d * d);
  if ((c & 31) == 0) red[c >> 5] = v;
  __syncthreads();
  float var = (red[0] + red[1] + red[2] + red[3] + red[4] + red[5] + red[6] + red[7]) * (1.0f / C_);
  float y = d * rsqrtf(var + EPS_) * g[c] + bt[c];
  float h = 0.5f * y * (1.0f + erff(y * 0.70710678118654752f));   // exact GELU
  H[(size_t)row * C_ + c] = f2bf(h);
}

// ---------- kernel 6: causal conv (K=31) as sliding-window WMMA GEMM + residual ----------
// 32 time-rows per block; each of 8 waves computes a 2Mx2N register tile
// (32x32 output), so every B fragment from L2 feeds 2 WMMAs per M-tile pair.
// Activation window is staged LDS-direct with CDNA5 async loads (ASYNCcnt).
__global__ __launch_bounds__(256) void k_conv(
    const __bf16* __restrict__ H, const __bf16* __restrict__ Wf,
    const float* __restrict__ cb, float* __restrict__ X) {
  const int b  = blockIdx.y;
  const int t0 = blockIdx.x * TMC;
  __shared__ __bf16 Hs[TMC + K_ - 1][C_ + 8];   // rows t0-30 .. t0+31

  int wave = threadIdx.x >> 5;
  int lane = threadIdx.x & 31;

  // causal zero-pad rows (only the first time-block has t<0 rows)
  if (t0 == 0) {
    for (int i = threadIdx.x; i < (K_ - 1) * (C_ / 4); i += 256) {
      int r  = i / (C_ / 4);
      int c4 = (i % (C_ / 4)) * 4;
      *(uint2*)&Hs[r][c4] = make_uint2(0u, 0u);
    }
  }
  // async global->LDS staging: one wave-wide b128 op moves a full 512B row.
  // Predicate (t >= 0) is wave-uniform, so EXEC is all-ones at the asm.
  for (int r = wave; r < TMC + K_ - 1; r += 8) {
    int t = t0 - (K_ - 1) + r;
    if (t >= 0) {
      unsigned ldsa = (unsigned)(uintptr_t)&Hs[r][lane * 8];
      const __bf16* gp = H + ((size_t)b * T_ + t) * C_ + lane * 8;
      asm volatile("global_load_async_to_lds_b128 %0, %1, off"
                   :: "v"(ldsa), "v"(gp) : "memory");
    }
  }
  asm volatile("s_wait_asynccnt 0x0" ::: "memory");
  __syncthreads();

  int half = lane >> 4;
  int nt0  = wave * 2;
  v8f acc00 = {}, acc01 = {}, acc10 = {}, acc11 = {};

  for (int k = 0; k < K_; ++k) {            // 31 taps
    const __bf16* wk = Wf + (size_t)k * C_ * C_;
#pragma unroll
    for (int kt = 0; kt < C_ / 32; ++kt) {  // reduction over c_in
      v16bf a0 = load_a_frag(&Hs[k][kt * 32], C_ + 8);        // rows m=0..15
      v16bf a1 = load_a_frag(&Hs[k + 16][kt * 32], C_ + 8);   // rows m=16..31
      v16bf b0 = load_b_frag(wk + ((size_t)(kt * 16 + nt0)) * 512);
      v16bf b1 = load_b_frag(wk + ((size_t)(kt * 16 + nt0 + 1)) * 512);
      acc00 = wmma_bf16(a0, b0, acc00);
      acc01 = wmma_bf16(a0, b1, acc01);
      acc10 = wmma_bf16(a1, b0, acc10);
      acc11 = wmma_bf16(a1, b1, acc11);
    }
  }

  int n0 = nt0 * 16 + (lane & 15);
  int n1 = n0 + 16;
  float c0 = cb[n0], c1 = cb[n1];
#pragma unroll
  for (int r = 0; r < 8; ++r) {
    int m = half * 8 + r;
    size_t o0 = ((size_t)b * T_ + t0 + m) * C_;
    size_t o1 = ((size_t)b * T_ + t0 + 16 + m) * C_;
    X[o0 + n0] = X[o0 + n0] + acc00[r] + c0;   // residual
    X[o0 + n1] = X[o0 + n1] + acc01[r] + c1;
    X[o1 + n0] = X[o1 + n0] + acc10[r] + c0;
    X[o1 + n1] = X[o1 + n1] + acc11[r] + c1;
  }
}

// ---------- kernel 7: final LayerNorm (in place on d_out) ----------
__global__ __launch_bounds__(256) void k_final_ln(
    float* __restrict__ X, const float* __restrict__ g, const float* __restrict__ bt) {
  int row = blockIdx.x;
  int c = threadIdx.x;
  float x = X[(size_t)row * C_ + c];
  __shared__ float red[8];
  float s = wave_sum(x);
  if ((c & 31) == 0) red[c >> 5] = s;
  __syncthreads();
  float mu = (red[0] + red[1] + red[2] + red[3] + red[4] + red[5] + red[6] + red[7]) * (1.0f / C_);
  __syncthreads();
  float d = x - mu;
  float v = wave_sum(d * d);
  if ((c & 31) == 0) red[c >> 5] = v;
  __syncthreads();
  float var = (red[0] + red[1] + red[2] + red[3] + red[4] + red[5] + red[6] + red[7]) * (1.0f / C_);
  X[(size_t)row * C_ + c] = d * rsqrtf(var + EPS_) * g[c] + bt[c];
}

// ---------- host launcher ----------
static inline size_t alignup(size_t x) { return (x + 255) & ~(size_t)255; }

extern "C" void kernel_launch(void* const* d_in, const int* in_sizes, int n_in,
                              void* d_out, int out_size, void* d_ws, size_t ws_size,
                              hipStream_t stream) {
  const float* pooled   = (const float*)d_in[0];
  const int*   dur      = (const int*)  d_in[1];
  const float* rel_pos  = (const float*)d_in[2];
  const float* W_in     = (const float*)d_in[3];
  const float* b_in     = (const float*)d_in[4];
  const float* W_pos    = (const float*)d_in[5];
  const float* b_pos    = (const float*)d_in[6];
  const float* ln_g     = (const float*)d_in[7];
  const float* ln_b     = (const float*)d_in[8];
  const float* conv_w   = (const float*)d_in[9];
  const float* conv_b   = (const float*)d_in[10];
  const float* out_g    = (const float*)d_in[11];
  const float* out_b    = (const float*)d_in[12];

  // workspace carve-up
  char* w = (char*)d_ws;
  int*    cum    = (int*)w;     w += alignup((size_t)B_ * N_ * sizeof(int));
  int*    idx    = (int*)w;     w += alignup((size_t)B_ * T_ * sizeof(int));
  __bf16* win_f  = (__bf16*)w;  w += alignup((size_t)DIN_ * C_ * sizeof(__bf16));
  __bf16* wpos_f = (__bf16*)w;  w += alignup((size_t)C_ * C_ * sizeof(__bf16));
  __bf16* conv_f = (__bf16*)w;  w += alignup((size_t)L_ * K_ * C_ * C_ * sizeof(__bf16));
  __bf16* H      = (__bf16*)w;  w += alignup((size_t)B_ * T_ * C_ * sizeof(__bf16));
  float*  X      = (float*)d_out;       // fp32 working buffer == output buffer

  // 0/1: duration -> frame index
  k_cumsum<<<B_, 32, 0, stream>>>(dur, cum);
  k_index<<<(B_ * T_ + 255) / 256, 256, 0, stream>>>(cum, idx);

  // 2/3: weight packing into WMMA B-fragment layout (bf16)
  k_pack<<<(DIN_ * C_ + 255) / 256, 256, 0, stream>>>(W_in, win_f, DIN_, C_);
  k_pack<<<(C_ * C_ + 255) / 256, 256, 0, stream>>>(W_pos, wpos_f, C_, C_);
  {
    int total = L_ * K_ * C_ * C_;
    k_pack_conv<<<(total + 255) / 256, 256, 0, stream>>>(conv_w, conv_f);
  }

  // 4: gather + in_proj + pos_proj
  dim3 s1_grid(T_ / TM, B_);
  k_stage1<<<s1_grid, 256, 0, stream>>>(pooled, idx, rel_pos,
                                        win_f, b_in, wpos_f, b_pos, X);

  // 5/6: L conv blocks
  dim3 conv_grid(T_ / TMC, B_);
  for (int l = 0; l < L_; ++l) {
    k_ln_gelu<<<B_ * T_, C_, 0, stream>>>(X, ln_g + l * C_, ln_b + l * C_, H);
    k_conv<<<conv_grid, 256, 0, stream>>>(H, conv_f + (size_t)l * K_ * C_ * C_,
                                          conv_b + l * C_, X);
  }

  // 7: final LayerNorm in place
  k_final_ln<<<B_ * T_, C_, 0, stream>>>(X, out_g, out_b);
}
